// residual_rnn_42649025249783
// MI455X (gfx1250) — matrix operand order repaired and checked by reading, not compiled, not run on hardware
//
#include <hip/hip_runtime.h>
#include <hip/hip_bf16.h>
#include <math.h>

// ---------------------------------------------------------------------------
// Types for CDNA5 WMMA (wave32, 16x16x32 bf16 -> f32)
// ---------------------------------------------------------------------------
typedef __bf16 bf16_t;
typedef bf16_t v16bf __attribute__((ext_vector_type(16)));
typedef bf16_t v8bf  __attribute__((ext_vector_type(8)));
typedef float  v8f   __attribute__((ext_vector_type(8)));

static __device__ __forceinline__ v16bf make_frag(const bf16_t* p0, const bf16_t* p1) {
    union { v16bf v; v8bf h[2]; } u;
    u.h[0] = *(const v8bf*)p0;
    u.h[1] = *(const v8bf*)p1;
    return u.v;
}

// Branch-free tanh: clamp + v_exp_f32 + v_rcp_f32. No EXEC-mask branches on
// the sequential critical path (OCML tanhf emits divergent range branches).
static __device__ __forceinline__ float fast_tanh(float x) {
    x = fminf(fmaxf(x, -9.0f), 9.0f);
    float e = __expf(2.0f * x);
    return (e - 1.0f) * __builtin_amdgcn_rcpf(e + 1.0f);
}

#define NB    64      // batch
#define NT    4096    // time steps
#define NH    256     // hidden
#define NI    7       // iterated features
#define NF    5       // fixed features
#define NXF   12      // I+F

#define HSTR  264     // lds h-tile row stride (bf16 elems, mult of 8)
#define ASTR  296     // mlp buffer A stride (>= 288, mult of 8)
#define BSTR  264     // mlp buffer B stride (>= 256, mult of 8)
#define TOK   32      // tokens per MLP workgroup (2 M-tiles)

// ---------------------------------------------------------------------------
// Kernel 0: convert an f32 weight [N][K] into zero-padded bf16 [Np][Kp]
// ---------------------------------------------------------------------------
__global__ void cvt_pad_kernel(const float* __restrict__ src, bf16_t* __restrict__ dst,
                               int N, int K, int Np, int Kp) {
    int idx = blockIdx.x * blockDim.x + threadIdx.x;
    if (idx >= Np * Kp) return;
    int n = idx / Kp, k = idx % Kp;
    float v = (n < N && k < K) ? src[n * K + k] : 0.0f;
    dst[idx] = (bf16_t)v;
}

// ---------------------------------------------------------------------------
// Kernel 1: fused input-projection + tanh RNN scan.
//   grid = 4 workgroups (16 batch rows each), block = 512 threads = 16 waves.
//   Wave w owns output columns [w*16, w*16+16); its B fragments of W_hh
//   (bf16) live in registers for the whole kernel. h and the x-stage tile
//   ping-pong in LDS -> exactly ONE workgroup barrier per time step.
// ---------------------------------------------------------------------------
__global__ __launch_bounds__(512) void rnn_scan_kernel(
    const float* __restrict__ x,
    const float* __restrict__ W_ih,
    const float* __restrict__ W_hh,
    const float* __restrict__ b_ih,
    const float* __restrict__ b_hh,
    bf16_t* __restrict__ ht,          // [B][T][H] bf16
    float*  __restrict__ h_last)      // [B][H] f32
{
    __shared__ bf16_t lds_h[2][16 * HSTR];
    __shared__ float  lds_x[2][16 * 8];

    const int tid  = threadIdx.x;
    const int wave = tid >> 5;
    const int lane = tid & 31;
    const int nl   = lane & 15;       // column within tile / A-matrix row
    const int half = lane >> 4;       // K half selector
    const int wg   = blockIdx.x;      // batch tile: rows wg*16 .. wg*16+15
    const int n    = wave * 16 + nl;  // global output column (0..255)

    // --- one-time: load this lane's B fragments of W_hh (bf16, registers) ---
    // B (32x16): lane holds column nl, K = kk*32 + half*8 .. +7 and kk*32 + 16 + half*8 .. +7.
    v16bf bfrag[8];
#pragma unroll
    for (int kk = 0; kk < 8; ++kk) {
        v16bf f;
#pragma unroll
        for (int e = 0; e < 16; ++e)
            f[e] = (bf16_t)W_hh[n * NH + kk * 32 + half * 8 + (e & 7) + (e >> 3) * 16];
        bfrag[kk] = f;
    }
    // per-lane input-projection constants
    float wih[NI];
#pragma unroll
    for (int i = 0; i < NI; ++i) wih[i] = W_ih[n * NI + i];
    const float bias = b_ih[n] + b_hh[n];

    // h0 = 0 and pre-stage x for t=0 into buffer 0
    for (int idx = tid; idx < 16 * HSTR; idx += 512) {
        lds_h[0][idx] = (bf16_t)0.0f;
        lds_h[1][idx] = (bf16_t)0.0f;
    }
    if (tid < 128) {
        int m = tid >> 3, i = tid & 7;
        if (i < NI)
            lds_x[0][m * 8 + i] = x[((size_t)(wg * 16 + m) * NT + 0) * NXF + i];
    }
    __syncthreads();

    float last[8];

    for (int t = 0; t < NT; ++t) {
        const int cur = t & 1, nxt = cur ^ 1;

        // prefetch x tile for step t+1 into the other buffer; the global-load
        // latency hides under this step's WMMA chain.
        if (t + 1 < NT && tid < 128) {
            int m = tid >> 3, i = tid & 7;
            if (i < NI)
                lds_x[nxt][m * 8 + i] =
                    x[((size_t)(wg * 16 + m) * NT + (t + 1)) * NXF + i];
        }

        // A (16x32 bf16): lane row = nl, chunks at kk*32+half*8 and +16.
        const bf16_t* aRow = &lds_h[cur][nl * HSTR + half * 8];
        v8f acc = {};
#pragma unroll
        for (int kk = 0; kk < 8; ++kk) {
            v16bf af = make_frag(aRow + kk * 32, aRow + kk * 32 + 16);
            acc = __builtin_amdgcn_wmma_f32_16x16x32_bf16(
                false, af, false, bfrag[kk], (short)0, acc, false, false);
        }

        // epilogue: + xW + bias, tanh, write next h + ht output
#pragma unroll
        for (int r = 0; r < 8; ++r) {
            const int m = r + 8 * half;           // row within batch tile
            float xw = bias;
#pragma unroll
            for (int i = 0; i < NI; ++i) xw += lds_x[cur][m * 8 + i] * wih[i];
            float v = fast_tanh(acc[r] + xw);
            last[r] = v;
            bf16_t vb = (bf16_t)v;
            lds_h[nxt][m * HSTR + n] = vb;
            ht[((size_t)(wg * 16 + m) * NT + t) * NH + n] = vb;
        }
        __syncthreads();   // publish h[nxt]/x[nxt]; retire reads of cur buffers
    }

#pragma unroll
    for (int r = 0; r < 8; ++r) {
        const int m = r + 8 * half;
        h_last[(wg * 16 + m) * NH + n] = last[r];
    }
}

// ---------------------------------------------------------------------------
// MLP layer helper: D = leaky(A * W^T + b); A in LDS (bf16), W padded bf16 in
// global (L2-resident). One (m-tile, n-tile) pair per wave iteration.
// ---------------------------------------------------------------------------
static __device__ __forceinline__ void mlp_layer(
    const bf16_t* __restrict__ in, int inStride,
    bf16_t* __restrict__ out, int outStride,
    const bf16_t* __restrict__ w, int Kp, int Ntiles, int Nreal,
    const float* __restrict__ bias, bool act,
    float* __restrict__ ot, int tok0,
    int wave, int nl, int half)
{
    const int Mtiles = TOK / 16;
    const int pairs  = Mtiles * Ntiles;
    for (int p = wave; p < pairs; p += 16) {
        const int ntile = p % Ntiles;
        const int mt    = p / Ntiles;
        const int n     = ntile * 16 + nl;
        const bf16_t* aRow = in + (mt * 16 + nl) * inStride + half * 8;
        const bf16_t* bRow = w + n * Kp + half * 8;
        v8f c = {};
        for (int kk = 0; kk < Kp; kk += 32) {
            v16bf af = make_frag(aRow + kk, aRow + kk + 16);
            v16bf bf = make_frag(bRow + kk, bRow + kk + 16);
            c = __builtin_amdgcn_wmma_f32_16x16x32_bf16(
                false, af, false, bf, (short)0, c, false, false);
        }
        const float bn = (n < Nreal) ? bias[n] : 0.0f;
#pragma unroll
        for (int r = 0; r < 8; ++r) {
            const int m = mt * 16 + r + 8 * half;
            float v = c[r] + bn;
            if (n >= Nreal) v = 0.0f;
            if (act) v = (v > 0.0f) ? v : 0.01f * v;
            if (out) {
                out[m * outStride + n] = (bf16_t)v;
            } else if (n < 7) {
                ot[(size_t)(tok0 + m) * 7 + n] = v;
            }
        }
    }
}

// ---------------------------------------------------------------------------
// Kernel 2: 7-layer MLP head over 32-token tiles.
//   grid = B*T/32 = 8192 workgroups, block = 512 threads = 16 waves.
// ---------------------------------------------------------------------------
__global__ __launch_bounds__(512) void mlp_kernel(
    const bf16_t* __restrict__ ht,    // [B*T][256] bf16
    const float*  __restrict__ x,     // [B][T][12] f32
    const bf16_t* __restrict__ gw,    // packed padded bf16 weights
    const float* __restrict__ b0, const float* __restrict__ b1,
    const float* __restrict__ b2, const float* __restrict__ b3,
    const float* __restrict__ b4, const float* __restrict__ b5,
    const float* __restrict__ b6,
    float* __restrict__ ot)           // [B*T][7] f32
{
    __shared__ bf16_t bufA[TOK * ASTR];
    __shared__ bf16_t bufB[TOK * BSTR];

    const int tid  = threadIdx.x;
    const int wave = tid >> 5;
    const int lane = tid & 31;
    const int nl   = lane & 15;
    const int half = lane >> 4;
    const int tok0 = blockIdx.x * TOK;

    // ---- stage z = [ht | x_fixed | zero-pad] into bufA (K padded to 288) ----
    {
        const uint4* src = (const uint4*)(ht + (size_t)tok0 * NH);
        for (int c = tid; c < TOK * 32; c += 512) {          // 32 x 16B per row
            int m = c >> 5, q = c & 31;
            *(uint4*)&bufA[m * ASTR + q * 8] = src[m * 32 + q];
        }
        for (int c = tid; c < TOK * 32; c += 512) {          // cols 256..287
            int m = c >> 5, j = c & 31;
            float v = 0.0f;
            if (j < NF) v = x[(size_t)(tok0 + m) * NXF + NI + j];
            bufA[m * ASTR + NH + j] = (bf16_t)v;
        }
    }
    __syncthreads();

    // packed weight offsets (elements)
    const bf16_t* W0p = gw;
    const bf16_t* W1p = gw + 73728;
    const bf16_t* W2p = gw + 106496;
    const bf16_t* W3p = gw + 114688;
    const bf16_t* W4p = gw + 116736;
    const bf16_t* W5p = gw + 117248;
    const bf16_t* W6p = gw + 117760;

    mlp_layer(bufA, ASTR, bufB, BSTR, W0p, 288, 16, 256, b0, true, nullptr, tok0, wave, nl, half);
    __syncthreads();
    mlp_layer(bufB, BSTR, bufA, ASTR, W1p, 256,  8, 128, b1, true, nullptr, tok0, wave, nl, half);
    __syncthreads();
    mlp_layer(bufA, ASTR, bufB, BSTR, W2p, 128,  4,  64, b2, true, nullptr, tok0, wave, nl, half);
    __syncthreads();
    mlp_layer(bufB, BSTR, bufA, ASTR, W3p,  64,  2,  32, b3, true, nullptr, tok0, wave, nl, half);
    __syncthreads();
    mlp_layer(bufA, ASTR, bufB, BSTR, W4p,  32,  1,  16, b4, true, nullptr, tok0, wave, nl, half);
    __syncthreads();
    // L5 consumes K_pad=32 but L4 produced only 16 cols -> zero cols 16..31
    for (int c = tid; c < TOK * 16; c += 512) {
        int m = c >> 4, j = c & 15;
        bufB[m * BSTR + 16 + j] = (bf16_t)0.0f;
    }
    __syncthreads();
    mlp_layer(bufB, BSTR, bufA, ASTR, W5p,  32,  1,   8, b5, true, nullptr, tok0, wave, nl, half);
    __syncthreads();
    for (int c = tid; c < TOK * 16; c += 512) {
        int m = c >> 4, j = c & 15;
        bufA[m * ASTR + 16 + j] = (bf16_t)0.0f;
    }
    __syncthreads();
    mlp_layer(bufA, ASTR, nullptr, 0, W6p,  32,  1,   7, b6, false, ot, tok0, wave, nl, half);
}

// ---------------------------------------------------------------------------
// Host launcher
// ---------------------------------------------------------------------------
extern "C" void kernel_launch(void* const* d_in, const int* in_sizes, int n_in,
                              void* d_out, int out_size, void* d_ws, size_t ws_size,
                              hipStream_t stream) {
    const float* x    = (const float*)d_in[0];
    const float* W_ih = (const float*)d_in[1];
    const float* W_hh = (const float*)d_in[2];
    const float* b_ih = (const float*)d_in[3];
    const float* b_hh = (const float*)d_in[4];
    const float* Wl[7], *bl[7];
    for (int j = 0; j < 7; ++j) {
        Wl[j] = (const float*)d_in[5 + 2 * j];
        bl[j] = (const float*)d_in[6 + 2 * j];
    }

    // workspace layout: [ ht bf16 (B*T*H) | packed padded bf16 weights ]
    bf16_t* ht = (bf16_t*)d_ws;
    bf16_t* gw = (bf16_t*)((char*)d_ws + (size_t)NB * NT * NH * sizeof(bf16_t));

    // weight conversion (zero-padded): {N, K, Np, Kp, elem offset}
    const int meta[7][5] = {
        {256, 261, 256, 288,      0},
        {128, 256, 128, 256,  73728},
        { 64, 128,  64, 128, 106496},
        { 32,  64,  32,  64, 114688},
        { 16,  32,  16,  32, 116736},
        {  8,  16,  16,  32, 117248},
        {  7,   8,  16,  32, 117760},
    };
    for (int j = 0; j < 7; ++j) {
        int total = meta[j][2] * meta[j][3];
        cvt_pad_kernel<<<(total + 255) / 256, 256, 0, stream>>>(
            Wl[j], gw + meta[j][4], meta[j][0], meta[j][1], meta[j][2], meta[j][3]);
    }

    float* ot     = (float*)d_out;
    float* h_last = (float*)d_out + (size_t)NB * NT * 7;

    rnn_scan_kernel<<<4, 512, 0, stream>>>(x, W_ih, W_hh, b_ih, b_hh, ht, h_last);

    mlp_kernel<<<(NB * NT) / TOK, 512, 0, stream>>>(
        ht, x, gw, bl[0], bl[1], bl[2], bl[3], bl[4], bl[5], bl[6], ot);
}
